// ProcessorNet_66228395705223
// MI455X (gfx1250) — matrix operand order repaired and checked by reading, not compiled, not run on hardware
//
#include <hip/hip_runtime.h>

typedef __attribute__((ext_vector_type(16))) __bf16    v16bf;
typedef __attribute__((ext_vector_type(2)))  __bf16    v2bf;
typedef __attribute__((ext_vector_type(8)))  float     v8f;
typedef __attribute__((ext_vector_type(8)))  unsigned  v8u;

#define WAVES_PER_BLOCK 8
#define THREADS (WAVES_PER_BLOCK * 32)

// pack two f32 into one dword of bf16 pair: single v_cvt_pk_bf16_f32
__device__ __forceinline__ unsigned pack_bf(float a, float b) {
  v2bf t;
  t[0] = (__bf16)a;
  t[1] = (__bf16)b;
  return __builtin_bit_cast(unsigned, t);
}

// lane-halves exchange on a packed word: lane L <-> L^16 (ds_swizzle SWAPX16)
__device__ __forceinline__ unsigned swap16u(unsigned x) {
  return (unsigned)__builtin_amdgcn_ds_swizzle((int)x, 0x401F);
}

// single-instruction relu (avoid fmax canonicalization double-max)
__device__ __forceinline__ float relu1(float x) {
  float r;
  asm("v_max_num_f32 %0, %1, 0" : "=v"(r) : "v"(x));
  return r;
}

__global__ __launch_bounds__(THREADS, 1)
void processor_net_kernel(const float* __restrict__ x,
                          const float* __restrict__ program,
                          const float* __restrict__ wd,
                          const float* __restrict__ bd,
                          const float* __restrict__ wu,
                          const float* __restrict__ bu,
                          float* __restrict__ out,
                          int n_rows)
{
  __shared__ float s_prog[64 * 16];
  for (int i = threadIdx.x; i < 64 * 16; i += THREADS) s_prog[i] = program[i];
  __syncthreads();

  const int lane = threadIdx.x & 31;
  const int lm   = lane & 15;        // M index within 16x16 tile / batch row
  const int half = lane >> 4;        // 0: lanes 0-15, 1: lanes 16-31
  const int koff = half << 3;        // A-operand K offset for this half-wave

  const long tile    = (long)blockIdx.x * WAVES_PER_BLOCK + (threadIdx.x >> 5);
  const long rowBase = tile * 16;
  if (rowBase >= n_rows) return;     // wave-uniform: EXEC stays all-ones
  const long row = rowBase + lm;

  // ---- GEMM1 A operands: wd[16,128], M=w=lm, K=H in 4 chunks of 32 ----
  v16bf a_wd[4];
#pragma unroll
  for (int j = 0; j < 4; ++j) {
    const float* p = wd + lm * 128 + 32 * j + 2 * koff;
    v8u aw;
#pragma unroll
    for (int v = 0; v < 4; ++v) aw[v]     = pack_bf(p[2 * v],      p[2 * v + 1]);
#pragma unroll
    for (int v = 0; v < 4; ++v) aw[4 + v] = pack_bf(p[8 + 2 * v], p[8 + 2 * v + 1]);
    a_wd[j] = __builtin_bit_cast(v16bf, aw);
  }

  // ---- GEMM2 A operands: wu[128,16], rows permuted so C-layout of the
  //      output matches the B-layout GEMM1 needs. K=16,17 carry bu (hi+lo). ----
  v16bf a_wu[8];
#pragma unroll
  for (int t = 0; t < 8; ++t) {
    int h = 32 * (t >> 1) + 8 * (t & 1) + (lm & 7) + ((lm >= 8) ? 16 : 0);
    const float* p = wu + h * 16 + koff;
    float bval = (half == 0) ? bu[h] : 0.0f;            // K=16,17 live in lanes 0-15
    float bhi  = (float)(__bf16)bval;
    float blo  = bval - bhi;
    v8u aw;
#pragma unroll
    for (int v = 0; v < 4; ++v) aw[v] = pack_bf(p[2 * v], p[2 * v + 1]);
    aw[4] = pack_bf(bhi, blo);                          // K=16,17
    aw[5] = 0u;
    aw[6] = 0u;
    aw[7] = 0u;
    a_wu[t] = __builtin_bit_cast(v16bf, aw);
  }

  // ---- bd broadcast matching reg^T C-layout ----
  float bdv[8];
#pragma unroll
  for (int r = 0; r < 8; ++r) bdv[r] = bd[r + koff];

  // ---- load x state as 8 f32 C-accumulators (S/S' row permutation) ----
  v8f acc[8];
  const float* xrow = x + row * 128;
#pragma unroll
  for (int t = 0; t < 8; ++t) {
    int hb = 32 * (t >> 1) + 8 * (t & 1) + 2 * koff;
    const float4* p = (const float4*)(xrow + hb);
    float4 lo = p[0], hi = p[1];
    v8f v = {lo.x, lo.y, lo.z, lo.w, hi.x, hi.y, hi.z, hi.w};
    acc[t] = v;
  }

  const unsigned ones_pk = 0x3F803F80u;   // packed bf16 {1.0, 1.0} for K=16,17 bias rows

  for (int l = 0; l < 64; ++l) {
    // ---- per-step scales, loaded early so ds latency hides under GEMM1 ----
    const float4* pp = (const float4*)(s_prog + l * 16 + koff);
    float4 p0 = pp[0], p1 = pp[1];

    // ---- build ALL four GEMM1 B operands as packed words (pure cvt_pk) ----
    v16bf bop[4];
#pragma unroll
    for (int j = 0; j < 4; ++j) {
      v8u bw;
#pragma unroll
      for (int v = 0; v < 4; ++v) bw[v]     = pack_bf(acc[2 * j][2 * v],     acc[2 * j][2 * v + 1]);
#pragma unroll
      for (int v = 0; v < 4; ++v) bw[4 + v] = pack_bf(acc[2 * j + 1][2 * v], acc[2 * j + 1][2 * v + 1]);
      bop[j] = __builtin_bit_cast(v16bf, bw);
    }

    // ---- GEMM1: reg^T = wd @ x^T + bd  (pure D->C accumulation chain) ----
    v8f c;
#pragma unroll
    for (int r = 0; r < 8; ++r) c[r] = bdv[r];
#pragma unroll
    for (int j = 0; j < 4; ++j) {
      c = __builtin_amdgcn_wmma_f32_16x16x32_bf16(false, a_wd[j], false, bop[j],
                                                  (short)0, c, false, false);
    }

    // ---- relu * program[l, w] ----
    float pv[8] = {p0.x, p0.y, p0.z, p0.w, p1.x, p1.y, p1.z, p1.w};
    float rg[8];
#pragma unroll
    for (int e = 0; e < 8; ++e) rg[e] = relu1(c[e]) * pv[e];

    // ---- GEMM2 B operand in word form:
    //      4 pair-packs, no swizzles; K=16,17 ones only in lanes 0-15 ----
    unsigned q[4];
#pragma unroll
    for (int u = 0; u < 4; ++u) q[u] = pack_bf(rg[2 * u], rg[2 * u + 1]);
    v8u brw;
#pragma unroll
    for (int u = 0; u < 4; ++u) {
      unsigned chi = (u == 0) ? ones_pk : 0u;           // lanes0-15: K=16,17 = 1.0 else 0
      brw[u]     = q[u];
      brw[4 + u] = (half == 0) ? chi : 0u;
    }
    v16bf br = __builtin_bit_cast(v16bf, brw);

    // ---- GEMM2: x^T += wu @ reg^T (+ bu via K=16,17), 8 independent WMMAs ----
#pragma unroll
    for (int t = 0; t < 8; ++t) {
      acc[t] = __builtin_amdgcn_wmma_f32_16x16x32_bf16(false, a_wu[t], false, br,
                                                       (short)0, acc[t], false, false);
    }
  }

  // ---- store result ----
  float* orow = out + row * 128;
#pragma unroll
  for (int t = 0; t < 8; ++t) {
    int hb = 32 * (t >> 1) + 8 * (t & 1) + 2 * koff;
    float4* p = (float4*)(orow + hb);
    float4 lo = {acc[t][0], acc[t][1], acc[t][2], acc[t][3]};
    float4 hi = {acc[t][4], acc[t][5], acc[t][6], acc[t][7]};
    p[0] = lo;
    p[1] = hi;
  }
}

extern "C" void kernel_launch(void* const* d_in, const int* in_sizes, int n_in,
                              void* d_out, int out_size, void* d_ws, size_t ws_size,
                              hipStream_t stream) {
  const float* x  = (const float*)d_in[0];
  const float* pg = (const float*)d_in[1];
  const float* wd = (const float*)d_in[2];
  const float* bd = (const float*)d_in[3];
  const float* wu = (const float*)d_in[4];
  const float* bu = (const float*)d_in[5];
  float* out = (float*)d_out;

  int n_rows = in_sizes[0] / 128;
  int tiles  = n_rows / 16;
  int blocks = (tiles + WAVES_PER_BLOCK - 1) / WAVES_PER_BLOCK;
  processor_net_kernel<<<blocks, THREADS, 0, stream>>>(x, pg, wd, bd, wu, bu, out, n_rows);
}
